// MultiHeadAttention_3058016715065
// MI455X (gfx1250) — matrix operand order repaired and checked by reading, not compiled, not run on hardware
//
#include <hip/hip_runtime.h>
#include <hip/hip_bf16.h>

typedef __bf16 bf16_t;
typedef __attribute__((ext_vector_type(16))) __bf16 v16bf;
typedef __attribute__((ext_vector_type(8)))  float  v8f;
typedef __attribute__((ext_vector_type(4)))  unsigned int u32x4;
typedef __attribute__((ext_vector_type(8)))  int i32x8;
typedef __attribute__((ext_vector_type(4)))  int i32x4;

#define D_MODEL 1024
#define NH      16
#define DK      64
#define S_LEN   2048
#define BATCH   2
#define NCHUNK  (S_LEN / 32)

__device__ __forceinline__ v8f wmma_bf16(v16bf a, v16bf b, v8f c) {
  // D = A(16x32 bf16) x B(32x16 bf16) + C(16x16 f32)
  return __builtin_amdgcn_wmma_f32_16x16x32_bf16(false, a, false, b,
                                                 (short)0, c, false, false);
}

// Max across the 16 lanes holding one C/D row (xor masks stay in each half).
__device__ __forceinline__ float half16_max(float v) {
  v = fmaxf(v, __shfl_xor(v, 1, 32));
  v = fmaxf(v, __shfl_xor(v, 2, 32));
  v = fmaxf(v, __shfl_xor(v, 4, 32));
  v = fmaxf(v, __shfl_xor(v, 8, 32));
  return v;
}

// Issue a 2D TDM copy: tile (tile_d1 rows x tile_d0 elems) of 2-byte data
// from global into LDS (packed, rows contiguous).  D# built per
// CDNA5 ISA (group0 + group1, groups 2/3 zero for 2D).
// This toolchain exposes the 6-arg clang-23 builtin:
//   (uint32x4 g0, int32x8 g1, int32x4, int32x4, int32x8, i32 cpol)
__device__ __forceinline__ void tdm_load_2d(const void* gaddr,
                                            const void* lds_ptr,
                                            unsigned tile_d0, unsigned tile_d1,
                                            unsigned tensor_d0, unsigned tensor_d1,
                                            unsigned long long stride0) {
  const unsigned long long ga = (unsigned long long)gaddr;
  // Generic LDS pointer: low 32 bits are the LDS byte offset.
  const unsigned lds_off = (unsigned)(unsigned long long)lds_ptr;

  u32x4 g0;
  g0[0] = 1u;                                        // count=1, user descriptor
  g0[1] = lds_off;                                   // lds_addr [63:32]
  g0[2] = (unsigned)(ga & 0xFFFFFFFFu);              // global_addr [95:64]
  g0[3] = (unsigned)((ga >> 32) & 0x01FFFFFFu)       // global_addr [120:96]
        | (2u << 30);                                // type=2 ("image")

  i32x8 g1;
  g1[0] = (int)(1u << 16);                           // data_size=1 (2 bytes)
  g1[1] = (int)((tensor_d0 & 0xFFFFu) << 16);        // tensor_dim0 [79:48]
  g1[2] = (int)((tensor_d0 >> 16) | ((tensor_d1 & 0xFFFFu) << 16)); // dim0 hi | dim1 lo
  g1[3] = (int)((tensor_d1 >> 16) | (tile_d0 << 16));               // dim1 hi | tile_dim0
  g1[4] = (int)(tile_d1 & 0xFFFFu);                  // tile_dim1 (tile_dim2=0)
  g1[5] = (int)(unsigned)(stride0 & 0xFFFFFFFFull);  // tensor_dim0_stride lo32
  g1[6] = (int)(unsigned)((stride0 >> 32) & 0xFFFFull); // stride hi16, dim1_stride=0
  g1[7] = 0;

  const i32x4 z4 = {0, 0, 0, 0};
  const i32x8 z8 = {0, 0, 0, 0, 0, 0, 0, 0};
  __builtin_amdgcn_tensor_load_to_lds(g0, g1, z4, z4, z8, 0);
}

// C[m,n] = sum_k X[m,k] * W[n,k] + bias[n]   (X @ W^T + b)
// MODE 0: bf16 out in [B, NH, S, DK]        (Q, K)
// MODE 2: bf16 out in [B, NH, DK, S]        (V transposed for PV fragments)
// MODE 1: f32 out row-major [M, N]          (final projection)
// One wave computes a 32x32 output tile (2x2 WMMA fragments).
template <int MODE>
__global__ __launch_bounds__(128) void gemm_xwT_kernel(
    const float* __restrict__ X, const float* __restrict__ W,
    const float* __restrict__ bias, void* __restrict__ out,
    int M, int N, int K)
{
  const int lane = threadIdx.x & 31;
  const int wave = threadIdx.x >> 5;
  const int h    = lane >> 4;
  const int l16  = lane & 15;

  const int tiles_n = N >> 5;
  const int tile = blockIdx.x * 4 + wave;
  const int m0 = (tile / tiles_n) << 5;
  const int n0 = (tile % tiles_n) << 5;

  const float* xr0 = X + (size_t)(m0 + l16) * K;
  const float* xr1 = X + (size_t)(m0 + 16 + l16) * K;
  const float* wr0 = W + (size_t)(n0 + l16) * K;
  const float* wr1 = W + (size_t)(n0 + 16 + l16) * K;

  v8f c[2][2] = {};

  for (int k0 = 0; k0 < K; k0 += 32) {
    // 32-byte vector loads; fp32 -> bf16 packs into v_cvt_pk_bf16_f32.
    const v8f xlo0 = *(const v8f*)(xr0 + k0 + 8 * h);       // A j=0..7
    const v8f xhi0 = *(const v8f*)(xr0 + k0 + 16 + 8 * h);  // A j=8..15
    const v8f xlo1 = *(const v8f*)(xr1 + k0 + 8 * h);
    const v8f xhi1 = *(const v8f*)(xr1 + k0 + 16 + 8 * h);
    const v8f wlo0 = *(const v8f*)(wr0 + k0 + 8 * h);       // B j=0..7
    const v8f whi0 = *(const v8f*)(wr0 + k0 + 16 + 8 * h);  // B j=8..15
    const v8f wlo1 = *(const v8f*)(wr1 + k0 + 8 * h);
    const v8f whi1 = *(const v8f*)(wr1 + k0 + 16 + 8 * h);

    v16bf a[2], b[2];
#pragma unroll
    for (int j = 0; j < 8; ++j) {
      a[0][j] = (__bf16)xlo0[j]; a[0][j + 8] = (__bf16)xhi0[j];
      a[1][j] = (__bf16)xlo1[j]; a[1][j + 8] = (__bf16)xhi1[j];
      b[0][j] = (__bf16)wlo0[j]; b[0][j + 8] = (__bf16)whi0[j];
      b[1][j] = (__bf16)wlo1[j]; b[1][j + 8] = (__bf16)whi1[j];
    }
#pragma unroll
    for (int ti = 0; ti < 2; ++ti)
#pragma unroll
      for (int tj = 0; tj < 2; ++tj)
        c[ti][tj] = wmma_bf16(a[ti], b[tj], c[ti][tj]);
  }

#pragma unroll
  for (int ti = 0; ti < 2; ++ti) {
#pragma unroll
    for (int tj = 0; tj < 2; ++tj) {
      const int n = n0 + 16 * tj + l16;
      const float bv = bias[n];
#pragma unroll
      for (int r = 0; r < 8; ++r) {
        const int m = m0 + 16 * ti + r + 8 * h;
        const float val = c[ti][tj][r] + bv;
        const int bb = m >> 11;             // m / S_LEN
        const int s  = m & (S_LEN - 1);
        const int hh = n >> 6;              // n / DK
        const int d  = n & (DK - 1);
        if (MODE == 0) {
          ((bf16_t*)out)[(((size_t)(bb * NH + hh) * S_LEN + s) * DK) + d] = (__bf16)val;
        } else if (MODE == 2) {
          ((bf16_t*)out)[(((size_t)(bb * NH + hh) * DK + d) * S_LEN) + s] = (__bf16)val;
        } else {
          ((float*)out)[(size_t)m * N + n] = val;
        }
      }
    }
  }
}

// Flash-attention workgroup: 4 waves share one (b,h); each wave owns a
// 16-row q-tile of a 64-row block.  K/V 32-key chunks are staged into LDS by
// the Tensor Data Mover (wave 0 issues, double-buffered, s_wait_tensorcnt +
// workgroup barriers), giving 4x reuse per global byte.
__global__ __launch_bounds__(128) void attn_kernel(
    const bf16_t* __restrict__ Q, const bf16_t* __restrict__ Km,
    const bf16_t* __restrict__ Vt, const int* __restrict__ mask,
    float* __restrict__ ctx)
{
  __shared__ __align__(128) bf16_t ldsK[2][32 * DK];  // 2 x 4 KB (rows=key)
  __shared__ __align__(128) bf16_t ldsV[2][DK * 32];  // 2 x 4 KB (rows=d)
  __shared__ __align__(16)  bf16_t ldsP[4][16 * 40];  // per-wave P strips

  const int lane = threadIdx.x & 31;
  const int wave = threadIdx.x >> 5;
  const int h    = lane >> 4;
  const int l16  = lane & 15;

  const int qb = blockIdx.x & ((S_LEN / 64) - 1);   // 32 q-blocks per head
  const int bh = blockIdx.x >> 5;
  const int b  = bh >> 4;
  const int hh = bh & (NH - 1);
  const int q0 = qb * 64 + wave * 16;

  const bf16_t* Qb = Q  + (size_t)bh * S_LEN * DK;
  const bf16_t* Kb = Km + (size_t)bh * S_LEN * DK;
  const bf16_t* Vb = Vt + (size_t)bh * DK * S_LEN;  // [DK][S]
  const int* maskb = mask + (size_t)b * S_LEN * S_LEN;

  // Q A-fragments (rows q0+l16, d-slices [0,32) and [32,64)), b128 loads.
  v16bf aq0, aq1;
  {
    const char* qp = (const char*)(Qb + (size_t)(q0 + l16) * DK);
    union { uint4 u[2]; v16bf v; } t0, t1;
    t0.u[0] = *(const uint4*)(qp + 16 * h);
    t0.u[1] = *(const uint4*)(qp + 32 + 16 * h);
    t1.u[0] = *(const uint4*)(qp + 64 + 16 * h);
    t1.u[1] = *(const uint4*)(qp + 96 + 16 * h);
    aq0 = t0.v; aq1 = t1.v;
  }

  v16bf ones;
#pragma unroll
  for (int j = 0; j < 16; ++j) ones[j] = (__bf16)1.0f;

  v8f acc[4] = {};
  float mrow[8], lrow[8];
#pragma unroll
  for (int r = 0; r < 8; ++r) { mrow[r] = -3.0e30f; lrow[r] = 0.0f; }

  bf16_t* P = &ldsP[wave][0];
  const float scale = 0.125f;      // 1/sqrt(DK)

  // TDM prologue: stage chunk 0 into buffer 0.
  if (wave == 0) {
    tdm_load_2d(Kb, &ldsK[0][0], DK, 32, DK, S_LEN, DK);
    tdm_load_2d(Vb, &ldsV[0][0], 32, DK, S_LEN, DK, S_LEN);
  }

  for (int i = 0; i < NCHUNK; ++i) {
    const int cur = i & 1;
    if (wave == 0) {
      if (i + 1 < NCHUNK) {
        const int kb1 = (i + 1) * 32;
        tdm_load_2d(Kb + (size_t)kb1 * DK, &ldsK[cur ^ 1][0], DK, 32, DK, S_LEN, DK);
        tdm_load_2d(Vb + kb1,              &ldsV[cur ^ 1][0], 32, DK, S_LEN, DK, S_LEN);
        __builtin_amdgcn_s_wait_tensorcnt((short)2);  // buffer `cur` complete
      } else {
        __builtin_amdgcn_s_wait_tensorcnt((short)0);
      }
    }
    __syncthreads();                                  // buffer `cur` visible

    const int kb0 = i * 32;
    const bf16_t* Klds = &ldsK[cur][0];
    const bf16_t* Vlds = &ldsV[cur][0];

    // Scores: two 16x16 tiles x two 32-wide d-slices.
    v8f s[2];
#pragma unroll
    for (int t = 0; t < 2; ++t) {
      const char* kr = (const char*)(Klds + (size_t)(16 * t + l16) * DK);
      union { uint4 u[2]; v16bf v; } b0, b1;
      b0.u[0] = *(const uint4*)(kr + 16 * h);
      b0.u[1] = *(const uint4*)(kr + 32 + 16 * h);
      b1.u[0] = *(const uint4*)(kr + 64 + 16 * h);
      b1.u[1] = *(const uint4*)(kr + 96 + 16 * h);
      v8f z = {};
      z = wmma_bf16(aq0, b0.v, z);
      s[t] = wmma_bf16(aq1, b1.v, z);
    }

    // Scale + mask (mask==0 -> -10000, matching reference).
#pragma unroll
    for (int t = 0; t < 2; ++t) {
      const int kp = kb0 + 16 * t + l16;
#pragma unroll
      for (int r = 0; r < 8; ++r) {
        const int qr = q0 + r + 8 * h;
        const int mv = maskb[(size_t)qr * S_LEN + kp];
        const float sv = s[t][r] * scale;
        s[t][r] = (mv == 0) ? -10000.0f : sv;
      }
    }

    // Online softmax: row max via lane reduction; keep corr for l update.
    float corr8[8];
#pragma unroll
    for (int r = 0; r < 8; ++r) {
      const float rm   = half16_max(fmaxf(s[0][r], s[1][r]));
      const float mnew = fmaxf(mrow[r], rm);
      const float corr = __expf(mrow[r] - mnew);
      s[0][r] = __expf(s[0][r] - mnew);
      s[1][r] = __expf(s[1][r] - mnew);
      corr8[r] = corr;
      mrow[r]  = mnew;
#pragma unroll
      for (int g = 0; g < 4; ++g) acc[g][r] *= corr;
    }

    // Relayout P (C/D form -> A form) through this wave's private LDS strip.
#pragma unroll
    for (int t = 0; t < 2; ++t)
#pragma unroll
      for (int r = 0; r < 8; ++r)
        P[(r + 8 * h) * 40 + 16 * t + l16] = (bf16_t)s[t][r];

    v16bf ap;
    {
      const char* prow = (const char*)(P + (size_t)l16 * 40);
      union { uint4 u[2]; v16bf v; } pu;
      pu.u[0] = *(const uint4*)(prow + 16 * h);
      pu.u[1] = *(const uint4*)(prow + 32 + 16 * h);
      ap = pu.v;
    }

    // Row sums via matrix pipe: rowsum(P) = P @ ones (replaces bpermutes).
    {
      v8f z = {};
      const v8f rs = wmma_bf16(ap, ones, z);
#pragma unroll
      for (int r = 0; r < 8; ++r) lrow[r] = lrow[r] * corr8[r] + rs[r];
    }

    // ctx += P(16x32) @ V(32x64); V B-fragments contiguous from transposed LDS tile.
#pragma unroll
    for (int g = 0; g < 4; ++g) {
      const char* vr = (const char*)(Vlds + (size_t)(16 * g + l16) * 32);
      union { uint4 u[2]; v16bf v; } bv;
      bv.u[0] = *(const uint4*)(vr + 16 * h);
      bv.u[1] = *(const uint4*)(vr + 32 + 16 * h);
      acc[g] = wmma_bf16(ap, bv.v, acc[g]);
    }

    __syncthreads();   // all waves done with buffer `cur` before TDM reuses it
  }

  // Normalize and scatter ctx into [B, S, D_MODEL] (column = h*DK + d).
#pragma unroll
  for (int r = 0; r < 8; ++r) {
    const float inv = 1.0f / lrow[r];
    const int srow = q0 + r + 8 * h;
#pragma unroll
    for (int g = 0; g < 4; ++g) {
      const int d = hh * DK + 16 * g + l16;
      ctx[((size_t)b * S_LEN + srow) * D_MODEL + d] = acc[g][r] * inv;
    }
  }
}

extern "C" void kernel_launch(void* const* d_in, const int* in_sizes, int n_in,
                              void* d_out, int out_size, void* d_ws, size_t ws_size,
                              hipStream_t stream) {
  (void)in_sizes; (void)n_in; (void)out_size; (void)ws_size;

  const float* q    = (const float*)d_in[0];
  const float* k    = (const float*)d_in[1];
  const float* v    = (const float*)d_in[2];
  const int*   mask = (const int*)d_in[3];
  const float* Wq_w = (const float*)d_in[4];
  const float* Wq_b = (const float*)d_in[5];
  const float* Wk_w = (const float*)d_in[6];
  const float* Wk_b = (const float*)d_in[7];
  const float* Wv_w = (const float*)d_in[8];
  const float* Wv_b = (const float*)d_in[9];
  const float* Wo_w = (const float*)d_in[10];
  const float* Wo_b = (const float*)d_in[11];

  const int M = BATCH * S_LEN;                 // 4096 rows
  const size_t nproj = (size_t)M * D_MODEL;    // 4M elements per projection

  // Workspace: Q/K [B,H,S,DK] bf16, V^T [B,H,DK,S] bf16 (3 x 8 MB),
  // ctx f32 (16 MB) = 40 MB total.
  bf16_t* wsQ  = (bf16_t*)d_ws;
  bf16_t* wsK  = wsQ + nproj;
  bf16_t* wsVt = wsK + nproj;
  float*  wsC  = (float*)(wsVt + nproj);

  dim3 blk(128);                                          // 4 waves (wave32)
  const int gemm_blocks = (M / 32) * (D_MODEL / 32) / 4;  // 1024
  gemm_xwT_kernel<0><<<gemm_blocks, blk, 0, stream>>>(q, Wq_w, Wq_b, wsQ,  M, D_MODEL, D_MODEL);
  gemm_xwT_kernel<0><<<gemm_blocks, blk, 0, stream>>>(k, Wk_w, Wk_b, wsK,  M, D_MODEL, D_MODEL);
  gemm_xwT_kernel<2><<<gemm_blocks, blk, 0, stream>>>(v, Wv_w, Wv_b, wsVt, M, D_MODEL, D_MODEL);

  const int attn_blocks = BATCH * NH * (S_LEN / 64);      // 1024
  attn_kernel<<<attn_blocks, blk, 0, stream>>>(wsQ, wsK, wsVt, mask, wsC);

  gemm_xwT_kernel<1><<<gemm_blocks, blk, 0, stream>>>(wsC, Wo_w, Wo_b, (float*)d_out, M, D_MODEL, D_MODEL);
}
